// MultiHeadAttention_Line_22050362098167
// MI455X (gfx1250) — matrix operand order repaired and checked by reading, not compiled, not run on hardware
//
#include <hip/hip_runtime.h>
#include <hip/hip_bf16.h>
#include <math.h>

typedef __attribute__((ext_vector_type(16))) _Float16 v16h;
typedef __attribute__((ext_vector_type(8)))  float    v8f;

// Problem geometry (fixed by reference): B=4, L=128, W=128, D=256, H=8, DK=32
#define NHEAD   8
#define DKH     32
#define NBL     512        // B*L
#define NROWS   65536      // B*L*W
#define DFEAT   256
#define OUT0_ELEMS 16777216   // NROWS*D  (fp32 out); attn (4096*128*128 fp32) follows

static __device__ __forceinline__ v8f wmma_f16(const v16h& a, const v16h& b, const v8f& c) {
  return __builtin_amdgcn_wmma_f32_16x16x32_f16(false, a, false, b, (short)0, c, false, false);
}

// ---------------------------------------------------------------- weights -> f16
__global__ void k_cvt_w(const float* __restrict__ Wq, const float* __restrict__ Wk,
                        const float* __restrict__ Wv, const float* __restrict__ Wfc,
                        _Float16* __restrict__ dst) {
  int i = blockIdx.x * blockDim.x + threadIdx.x;      // 65536 elems / matrix
  dst[i]          = (_Float16)Wq[i];
  dst[i +  65536] = (_Float16)Wk[i];
  dst[i + 131072] = (_Float16)Wv[i];
  dst[i + 196608] = (_Float16)Wfc[i];
}

// ---------------------------------------------------------------- QKV projection
// Y = X @ W^T + b, written f16 head-major [bl][h][w][dk]; Q pre-scaled by 1/sqrt(DK).
// Block = 8 waves covering one 16-row tile x 128 cols; output staged via LDS for
// coalesced 128b stores.
__global__ __launch_bounds__(256) void k_proj(
    const float* __restrict__ q, const float* __restrict__ k, const float* __restrict__ v,
    const _Float16* __restrict__ w16,
    const float* __restrict__ bq, const float* __restrict__ bk, const float* __restrict__ bv,
    _Float16* __restrict__ qh, _Float16* __restrict__ kh, _Float16* __restrict__ vh) {
  __shared__ _Float16 tile[16 * 128];   // 16 rows x 128 cols of Y (f16), 4KB

  const int tid  = threadIdx.x;
  const int lane = tid & 31;
  const int wv   = tid >> 5;
  const int gw0  = blockIdx.x * 8;          // first wave-tile of this block
  const int sel  = gw0 >> 16;               // 0:q 1:k 2:v (uniform in block)
  const int rem0 = gw0 & 65535;
  const int mt   = rem0 >> 4;               // row tile (uniform in block)
  const int nt0  = rem0 & 15;               // 0 or 8 (uniform in block)
  const int m0   = mt << 4;
  const int n0   = (nt0 + wv) << 4;         // this wave's col tile

  const float*    X    = (sel == 0) ? q  : (sel == 1) ? k  : v;
  const float*    bias = (sel == 0) ? bq : (sel == 1) ? bk : bv;
  const _Float16* Wt   = w16 + (size_t)sel * 65536;
  _Float16*       Y    = (sel == 0) ? qh : (sel == 1) ? kh : vh;

  const int m   = lane & 15;
  const int n   = lane & 15;
  const int mb  = (lane >> 4) ? 8 : 0;
  const int kbA = (lane >> 4) ? 8 : 0;
  const int kbB = (lane >> 4) ? 8 : 0;

  v8f c;
  const float bval = bias[n0 + n];
  #pragma unroll
  for (int r = 0; r < 8; ++r) c[r] = bval;

  for (int kk = 0; kk < 256; kk += 32) {
    v16h a, b;
    const float* Xr = X + (size_t)(m0 + m) * DFEAT + kk + kbA;
    if (kk + 32 < 256) __builtin_prefetch(Xr + 32, 0, 1);   // global_prefetch_b8
    #pragma unroll
    for (int vv = 0; vv < 8; ++vv) {           // ISA 16-bit A 16x32 layout
      int k0 = ((vv & 4) ? 16 : 0) + (vv & 3) * 2;
      a[2*vv]   = (_Float16)Xr[k0];
      a[2*vv+1] = (_Float16)Xr[k0+1];
    }
    const _Float16* Wr = Wt + (size_t)(n0 + n) * DFEAT + kk + kbB;  // B[k][n] = W[n][k]
    #pragma unroll
    for (int i = 0; i < 16; ++i) b[i] = Wr[(i < 8) ? i : i + 8];
    c = wmma_f16(a, b, c);
  }

  // stage this wave's 16x16 D tile into the block's 16x128 LDS tile
  const float rscale = (sel == 0) ? 0.17677669529663687f : 1.0f;   // 1/sqrt(32)
  #pragma unroll
  for (int r = 0; r < 8; ++r)
    tile[(mb + r) * 128 + wv * 16 + n] = (_Float16)(c[r] * rscale);
  __syncthreads();

  // coalesced store: region = 4 heads x 16 rows x 32 dk (64B contiguous chunks)
  {
    const int bl  = m0 >> 7;
    const int w0  = m0 & 127;
    const int h00 = nt0 >> 1;          // 0 or 4
    const int chunk = tid >> 2;        // 0..63
    const int part  = tid & 3;         // 0..3 (uint4 = 8 halves)
    const int h   = chunk >> 4;        // 0..3
    const int w16r = chunk & 15;       // 0..15
    const uint4* src = (const uint4*)(tile + w16r * 128 + h * 32) + part;
    uint4* dst = (uint4*)(Y + (((size_t)bl * NHEAD + h00 + h) * 128 + (w0 + w16r)) * DKH) + part;
    *dst = *src;
  }
}

// ---------------------------------------------------------------- fused attention
// one workgroup per (b,l,h); S=QK^T -> softmax -> O=PV, S/P stay in LDS
__global__ __launch_bounds__(256) void k_attn(
    const _Float16* __restrict__ qh, const _Float16* __restrict__ kh,
    const _Float16* __restrict__ vh, float* __restrict__ attn_out,
    _Float16* __restrict__ ofull) {
  __shared__ _Float16 Qs[128 * 32];      // Q tile; reused as O staging later
  __shared__ _Float16 Ks[128 * 32];
  __shared__ _Float16 Vs[128 * 32];
  __shared__ _Float16 Ps[128 * 128];     // logits then probs (f16) — 56KB total LDS

  const int tid  = threadIdx.x;
  const int lane = tid & 31;
  const int wv   = tid >> 5;             // 8 waves
  const int blk  = blockIdx.x;           // bl*8 + h
  const size_t base = (size_t)blk * (128 * 32);

  {  // cooperative tile load: 3 x 8KB, 128b vector copies
    const uint4* gq = (const uint4*)(qh + base);
    const uint4* gk = (const uint4*)(kh + base);
    const uint4* gv = (const uint4*)(vh + base);
    uint4* sq = (uint4*)Qs; uint4* sk = (uint4*)Ks; uint4* sv = (uint4*)Vs;
    for (int i = tid; i < 512; i += 256) { sq[i] = gq[i]; sk[i] = gk[i]; sv[i] = gv[i]; }
  }
  __syncthreads();

  {  // S = Q * K^T  (K-dim = 32, single WMMA per tile); wave wv owns M-tile wv
    const int m0  = wv << 4;
    const int m   = lane & 15, n = lane & 15;
    const int mb  = (lane >> 4) ? 8 : 0;
    const int kbA = (lane >> 4) ? 8 : 0;
    const int kbB = (lane >> 4) ? 8 : 0;
    v16h a;
    const _Float16* Qr = Qs + (m0 + m) * 32 + kbA;
    #pragma unroll
    for (int vv = 0; vv < 8; ++vv) {
      int k0 = ((vv & 4) ? 16 : 0) + (vv & 3) * 2;
      a[2*vv] = Qr[k0]; a[2*vv+1] = Qr[k0+1];
    }
    #pragma unroll
    for (int nt = 0; nt < 8; ++nt) {
      const int n0 = nt << 4;
      v16h b;
      const _Float16* Kr = Ks + (n0 + n) * 32 + kbB;   // B[k][n] = K[n][k]
      #pragma unroll
      for (int i = 0; i < 16; ++i) b[i] = Kr[(i < 8) ? i : i + 8];
      v8f c = {};
      c = wmma_f16(a, b, c);
      #pragma unroll
      for (int r = 0; r < 8; ++r) Ps[(m0 + mb + r) * 128 + n0 + n] = (_Float16)c[r];
    }
  }
  __syncthreads();

  {  // wave-parallel softmax: wave owns 16 rows; 32 lanes x 4 cols per row.
     // fp32 attn output written as coalesced float4; normalized P kept f16 in LDS.
    for (int i = 0; i < 16; ++i) {
      const int r = wv * 16 + i;
      _Float16* Pr = Ps + r * 128 + lane * 4;
      float x0 = (float)Pr[0], x1 = (float)Pr[1], x2 = (float)Pr[2], x3 = (float)Pr[3];
      float mx = fmaxf(fmaxf(x0, x1), fmaxf(x2, x3));
      #pragma unroll
      for (int o = 16; o >= 1; o >>= 1) mx = fmaxf(mx, __shfl_xor(mx, o, 32));
      float e0 = __expf(x0 - mx), e1 = __expf(x1 - mx);
      float e2 = __expf(x2 - mx), e3 = __expf(x3 - mx);
      float s = e0 + e1 + e2 + e3;
      #pragma unroll
      for (int o = 16; o >= 1; o >>= 1) s += __shfl_xor(s, o, 32);
      const float inv = 1.0f / s;
      e0 *= inv; e1 *= inv; e2 *= inv; e3 *= inv;
      float4 p4 = make_float4(e0, e1, e2, e3);
      *(float4*)(attn_out + ((size_t)blk * 128 + r) * 128 + lane * 4) = p4;
      Pr[0] = (_Float16)e0; Pr[1] = (_Float16)e1;
      Pr[2] = (_Float16)e2; Pr[3] = (_Float16)e3;
    }
  }
  __syncthreads();

  {  // O = P * V : [128x128]x[128x32], K loop 4x32; stage O in LDS (reuse Qs)
    _Float16* Os = Qs;                       // Q is dead after S phase
    const int m0  = wv << 4;
    const int m   = lane & 15, nl = lane & 15;
    const int mb  = (lane >> 4) ? 8 : 0;
    const int kbA = (lane >> 4) ? 8 : 0;
    const int kbB = (lane >> 4) ? 8 : 0;
    #pragma unroll
    for (int nt = 0; nt < 2; ++nt) {
      const int n0 = nt << 4;
      v8f c = {};
      for (int kk = 0; kk < 128; kk += 32) {
        v16h a, b;
        const _Float16* Pr = Ps + (m0 + m) * 128 + kk + kbA;
        #pragma unroll
        for (int vv = 0; vv < 8; ++vv) {
          int k0 = ((vv & 4) ? 16 : 0) + (vv & 3) * 2;
          a[2*vv] = Pr[k0]; a[2*vv+1] = Pr[k0+1];
        }
        const _Float16* Vr = Vs + (kk + kbB) * 32 + n0 + nl;  // B row-major [k][n]
        #pragma unroll
        for (int i = 0; i < 16; ++i) b[i] = Vr[((i < 8) ? i : i + 8) * 32];
        c = wmma_f16(a, b, c);
      }
      #pragma unroll
      for (int r = 0; r < 8; ++r)
        Os[(m0 + mb + r) * 32 + n0 + nl] = (_Float16)c[r];
    }
    __syncthreads();
    // coalesced O store: 128 rows x 32 dk -> Ofull[row][h*32..], 64B chunks
    const int bl = blk >> 3, h = blk & 7;
    const uint4* src = (const uint4*)Os;
    for (int i = tid; i < 512; i += 256) {
      const int row  = i >> 2;
      const int part = i & 3;
      uint4* dst = (uint4*)(ofull + ((size_t)bl * 128 + row) * DFEAT + h * 32) + part;
      *dst = src[i];
    }
  }
}

// ---------------------------------------------------------------- FC + residual + LayerNorm
__global__ __launch_bounds__(128) void k_fc_ln(
    const _Float16* __restrict__ ofull, const _Float16* __restrict__ wfc16,
    const float* __restrict__ bfc, const float* __restrict__ resid,
    const float* __restrict__ ln_w, const float* __restrict__ ln_b,
    float* __restrict__ out) {
  __shared__ float Yt[16 * 256];
  __shared__ float red[16 * 8 * 2];
  __shared__ float stats[16 * 2];

  const int tid  = threadIdx.x;
  const int lane = tid & 31;
  const int wv   = tid >> 5;             // 4 waves
  const int r0   = blockIdx.x * 16;      // 16 rows per block

  const int m   = lane & 15, n = lane & 15;
  const int mb  = (lane >> 4) ? 8 : 0;
  const int kbA = (lane >> 4) ? 8 : 0;
  const int kbB = (lane >> 4) ? 8 : 0;

  for (int nt = wv; nt < 16; nt += 4) {
    const int n0 = nt << 4;
    v8f c;
    const float bval = bfc[n0 + n];
    #pragma unroll
    for (int r = 0; r < 8; ++r) c[r] = bval;
    for (int kk = 0; kk < 256; kk += 32) {
      v16h a, b;
      const _Float16* Ar = ofull + (size_t)(r0 + m) * DFEAT + kk + kbA;
      #pragma unroll
      for (int vv = 0; vv < 8; ++vv) {
        int k0 = ((vv & 4) ? 16 : 0) + (vv & 3) * 2;
        a[2*vv] = Ar[k0]; a[2*vv+1] = Ar[k0+1];
      }
      const _Float16* Br = wfc16 + (size_t)(n0 + n) * DFEAT + kk + kbB;
      #pragma unroll
      for (int i = 0; i < 16; ++i) b[i] = Br[(i < 8) ? i : i + 8];
      c = wmma_f16(a, b, c);
    }
    #pragma unroll
    for (int r = 0; r < 8; ++r) {
      int mm = mb + r;
      Yt[mm * 256 + n0 + n] = c[r] + resid[(size_t)(r0 + mm) * DFEAT + n0 + n];
    }
  }
  __syncthreads();

  // LayerNorm: 8 threads per row (32 cols each), eps=1e-6
  const int row = tid >> 3;
  const int seg = tid & 7;
  {
    float s = 0.f, s2 = 0.f;
    const float* y = Yt + row * 256 + seg * 32;
    #pragma unroll
    for (int j = 0; j < 32; ++j) { float v0 = y[j]; s += v0; s2 += v0 * v0; }
    red[(row * 8 + seg) * 2 + 0] = s;
    red[(row * 8 + seg) * 2 + 1] = s2;
  }
  __syncthreads();
  if (seg == 0) {
    float s = 0.f, s2 = 0.f;
    #pragma unroll
    for (int j = 0; j < 8; ++j) { s += red[(row * 8 + j) * 2]; s2 += red[(row * 8 + j) * 2 + 1]; }
    float mu  = s * (1.0f / 256.0f);
    float var = s2 * (1.0f / 256.0f) - mu * mu;
    stats[row * 2 + 0] = mu;
    stats[row * 2 + 1] = rsqrtf(var + 1e-6f);
  }
  __syncthreads();
  {
    const float mu = stats[row * 2], rs = stats[row * 2 + 1];
    const float* y  = Yt + row * 256 + seg * 32;
    float*       o  = out + (size_t)(r0 + row) * DFEAT + seg * 32;
    const float* lw = ln_w + seg * 32;
    const float* lb = ln_b + seg * 32;
    #pragma unroll
    for (int j = 0; j < 32; ++j) o[j] = (y[j] - mu) * rs * lw[j] + lb[j];
  }
}

// ---------------------------------------------------------------- host launcher
extern "C" void kernel_launch(void* const* d_in, const int* in_sizes, int n_in,
                              void* d_out, int out_size, void* d_ws, size_t ws_size,
                              hipStream_t stream) {
  const float* q    = (const float*)d_in[0];
  const float* k    = (const float*)d_in[1];
  const float* v    = (const float*)d_in[2];
  const float* Wq   = (const float*)d_in[3];
  const float* bq   = (const float*)d_in[4];
  const float* Wk   = (const float*)d_in[5];
  const float* bk   = (const float*)d_in[6];
  const float* Wv   = (const float*)d_in[7];
  const float* bv   = (const float*)d_in[8];
  const float* Wfc  = (const float*)d_in[9];
  const float* bfc  = (const float*)d_in[10];
  const float* ln_w = (const float*)d_in[11];
  const float* ln_b = (const float*)d_in[12];

  float* out  = (float*)d_out;
  float* attn = out + OUT0_ELEMS;

  // workspace: [4x 256x256 f16 weights][Qh][Kh][Vh][O] f16 head-major, ~128.5 MB
  _Float16* w16 = (_Float16*)d_ws;
  _Float16* qh  = w16 + 262144;
  _Float16* kh  = qh + OUT0_ELEMS;
  _Float16* vh  = kh + OUT0_ELEMS;
  _Float16* of  = vh + OUT0_ELEMS;

  k_cvt_w<<<256, 256, 0, stream>>>(Wq, Wk, Wv, Wfc, w16);
  // 3 mats * 4096 Mtiles * 16 Ntiles = 196608 wave-tiles / 8 waves per block
  k_proj<<<24576, 256, 0, stream>>>(q, k, v, w16, bq, bk, bv, qh, kh, vh);
  // one workgroup per (b,l,h)
  k_attn<<<4096, 256, 0, stream>>>(qh, kh, vh, attn, of);
  // 16 rows per block
  k_fc_ln<<<4096, 128, 0, stream>>>(of, w16 + 196608, bfc, q, ln_w, ln_b, out);
}
